// CausalSelfAttention_22127671509246
// MI455X (gfx1250) — matrix operand order repaired and checked by reading, not compiled, not run on hardware
//
#include <hip/hip_runtime.h>
#include <hip/hip_bf16.h>

typedef __attribute__((ext_vector_type(16))) _Float16 v16h;
typedef __attribute__((ext_vector_type(4)))  _Float16 v4h;
typedef __attribute__((ext_vector_type(8)))  float    v8f;
typedef __attribute__((ext_vector_type(4)))  float    f32x4;
typedef __attribute__((ext_vector_type(4)))  unsigned int u32x4;
typedef int v4i __attribute__((vector_size(16)));   // matches async-LDS builtin param

#define BATCH   2
#define SEQ     2048
#define D_MODEL 1024
#define N_HEADS 16
#define HD      64

#define ASG __attribute__((address_space(1)))
#define ASL __attribute__((address_space(3)))

#if defined(__gfx1250__) && \
    __has_builtin(__builtin_amdgcn_global_load_async_to_lds_b128) && \
    __has_builtin(__builtin_amdgcn_s_wait_asynccnt)
#define HAVE_ASYNC 1
#endif

static __device__ __forceinline__ v8f wmma_f16(v16h a, v16h b, v8f c) {
    // D = A(16x32 f16) * B(32x16 f16) + C(16x16 f32)
    return __builtin_amdgcn_wmma_f32_16x16x32_f16(false, a, false, b,
                                                  (short)0, c, false, false);
}

// ---------------------------------------------------------------------------
// Kernel 1: fused QKV projection.  Y = x @ W + b, output converted to f16 and
// stored head-major [B, H, S, 64].  grid = (16 heads, 32 row-tiles, 3 qkv),
// block = 256 threads = 8 waves.  Block tile: 128 rows x 64 cols (one head).
// Each wave: 16 rows x 64 cols = 4 C fragments, 4 wmma per 32-deep K step.
// Staging: b128 global loads gathered into registers first (one wait), then
// f16 conversion; x tile written with 8B ds_store_b64 packs.
// ---------------------------------------------------------------------------
__global__ __launch_bounds__(256)
void qkv_proj_kernel(const float* __restrict__ x,
                     const float* __restrict__ Wq, const float* __restrict__ bq,
                     const float* __restrict__ Wk, const float* __restrict__ bk,
                     const float* __restrict__ Wv, const float* __restrict__ bv,
                     _Float16* __restrict__ q_ws, _Float16* __restrict__ k_ws,
                     _Float16* __restrict__ v_ws)
{
    __shared__ _Float16 lds_a[128 * 36];   // x tile [row][k], stride 36 (8B-aligned rows)
    __shared__ _Float16 lds_b[64 * 36];    // W tile [n][k],   stride 36

    const int t    = threadIdx.x;
    const int wave = t >> 5;
    const int lane = t & 31;
    const int hlf  = lane >> 4;            // which 16-lane half of the wave
    const int l16  = lane & 15;

    const int h     = blockIdx.x;          // head == 64-wide N tile
    const int row0  = blockIdx.y * 128;    // global row base (over B*S)
    const int which = blockIdx.z;          // 0=Q 1=K 2=V

    const float* W    = (which == 0) ? Wq : (which == 1) ? Wk : Wv;
    const float* bias = (which == 0) ? bq : (which == 1) ? bk : bv;
    _Float16*    outp = (which == 0) ? q_ws : (which == 1) ? k_ws : v_ws;

    v8f acc[4] = {};

    for (int k0 = 0; k0 < D_MODEL; k0 += 32) {
        // ---- gather all global data for this K-step into registers first ----
        f32x4 xv[4];   // x tile: 128x32 floats = 1024 float4, 4 per thread
        int   xr[4], xc[4];
        #pragma unroll
        for (int e = 0; e < 4; ++e) {
            int idx = t + e * 256;                  // 0..1023 (float4 chunks)
            xr[e] = idx >> 3;                       // row 0..127
            xc[e] = (idx & 7) * 4;                  // col 0,4,..,28
            xv[e] = *(const f32x4*)&x[(size_t)(row0 + xr[e]) * D_MODEL + k0 + xc[e]];
        }
        f32x4 wv[2];   // W tile: 32x64 floats = 512 float4, 2 per thread
        int   wk[2], wn[2];
        #pragma unroll
        for (int e = 0; e < 2; ++e) {
            int idx = t + e * 256;                  // 0..511
            wk[e] = idx >> 4;                       // k 0..31
            wn[e] = (idx & 15) * 4;                 // n 0,4,..,60
            wv[e] = *(const f32x4*)&W[(size_t)(k0 + wk[e]) * D_MODEL + h * 64 + wn[e]];
        }
        // ---- convert + store to LDS ----
        #pragma unroll
        for (int e = 0; e < 4; ++e) {
            v4h p;
            p[0] = (_Float16)xv[e][0]; p[1] = (_Float16)xv[e][1];
            p[2] = (_Float16)xv[e][2]; p[3] = (_Float16)xv[e][3];
            *(v4h*)&lds_a[xr[e] * 36 + xc[e]] = p;  // 8B aligned: 72*r + 8*c4
        }
        #pragma unroll
        for (int e = 0; e < 2; ++e) {
            #pragma unroll
            for (int q = 0; q < 4; ++q)
                lds_b[(wn[e] + q) * 36 + wk[e]] = (_Float16)wv[e][q];
        }
        // prefetch next K-step tiles into cache (global_prefetch_b8)
        if (k0 + 32 < D_MODEL) {
            __builtin_prefetch(
                &x[(size_t)(row0 + (t >> 1)) * D_MODEL + k0 + 32 + (t & 1) * 16],
                0, 3);
            if (t < 128)
                __builtin_prefetch(
                    &W[(size_t)(k0 + 32 + (t >> 2)) * D_MODEL + h * 64 + (t & 3) * 16],
                    0, 3);
        }
        __syncthreads();

        // A fragment: lane row = wave*16 + l16; contiguous 16 K per half-wave
        v16h af;
        {
            const _Float16* ap = &lds_a[(wave * 16 + l16) * 36];
            const int klo = hlf * 16;
            #pragma unroll
            for (int i = 0; i < 8; ++i) af[i]     = ap[klo + i];
            #pragma unroll
            for (int i = 0; i < 8; ++i) af[8 + i] = ap[klo + 8 + i];
        }
        #pragma unroll
        for (int nt = 0; nt < 4; ++nt) {
            v16h bf;
            const _Float16* bp = &lds_b[(nt * 16 + l16) * 36 + hlf * 16];
            #pragma unroll
            for (int i = 0; i < 16; ++i) bf[i] = bp[i];
            acc[nt] = wmma_f16(af, bf, acc[nt]);
        }
        __syncthreads();
    }

    // epilogue: + bias, convert to f16, store head-major [B,H,S,64]
    #pragma unroll
    for (int nt = 0; nt < 4; ++nt) {
        const float bb = bias[h * 64 + nt * 16 + l16];
        #pragma unroll
        for (int i = 0; i < 8; ++i) {
            int M = hlf * 8 + i;
            int r = row0 + wave * 16 + M;               // global row in B*S
            int b = r >> 11;                            // / 2048
            int s = r & 2047;
            float val = acc[nt][i] + bb;
            outp[(((size_t)(b * N_HEADS + h) * SEQ + s) * HD)
                 + nt * 16 + l16] = (_Float16)val;
        }
    }
}

// ---------------------------------------------------------------------------
// Kernel 2: attention (softmax over ALL keys; reference never applies mask).
// grid = (S/128 query tiles, B*H), block = 256 = 8 waves.  Each wave owns 16
// query rows.  Flash-attention loop over 64-key tiles: S=Q*K^T (WMMA),
// online softmax (half-wave shfl_xor reductions), O += P*V (WMMA).
// K tile staged via GLOBAL_LOAD_ASYNC_TO_LDS_B128 (ASYNCcnt); V tile staged
// with b128 register gathers then transposed b16 LDS stores.
// ---------------------------------------------------------------------------
__global__ __launch_bounds__(256)
void attn_kernel(const _Float16* __restrict__ q_ws,
                 const _Float16* __restrict__ k_ws,
                 const _Float16* __restrict__ v_ws,
                 float* __restrict__ out)
{
    __shared__ _Float16 lds_k[64 * 72];        // K tile [j][d]  (== B of Q*K^T)
    __shared__ _Float16 lds_v[64 * 72];        // V tile [d][j]  (transposed)
    __shared__ _Float16 lds_p[8 * 16 * 72];    // per-wave P staging [m][j]

    const int t    = threadIdx.x;
    const int wave = t >> 5;
    const int lane = t & 31;
    const int hlf  = lane >> 4;
    const int l16  = lane & 15;

    const int bh = blockIdx.y;                 // 0..31
    const int b  = bh >> 4;
    const int h  = bh & 15;
    const int q0 = blockIdx.x * 128;
    const size_t base = (size_t)bh * SEQ * HD;

    // Q fragments (hd=64 -> two 32-deep K steps), pre-scaled by 1/sqrt(64)
    v16h aq[2];
    {
        const int srow = q0 + wave * 16 + l16;
        const _Float16* qp = q_ws + base + (size_t)srow * HD;
        const _Float16 sc = (_Float16)0.125f;
        #pragma unroll
        for (int ks = 0; ks < 2; ++ks) {
            const int klo = ks * 32 + hlf * 16;
            #pragma unroll
            for (int i = 0; i < 8; ++i) aq[ks][i]     = qp[klo + i] * sc;
            #pragma unroll
            for (int i = 0; i < 8; ++i) aq[ks][8 + i] = qp[klo + 8 + i] * sc;
        }
    }

    float m[8], l[8];
    #pragma unroll
    for (int i = 0; i < 8; ++i) { m[i] = -__builtin_inff(); l[i] = 0.0f; }
    v8f o[4] = {};

    for (int j0 = 0; j0 < SEQ; j0 += 64) {
        // ---- stage K tile [j][d]: async global->LDS DMA copy ----
#if defined(HAVE_ASYNC)
        #pragma unroll
        for (int e = 0; e < 2; ++e) {
            int idx = t + e * 256;                 // 0..511 (16B chunks)
            int j = idx >> 3, c = idx & 7;
            __builtin_amdgcn_global_load_async_to_lds_b128(
                (ASG v4i*)(k_ws + base + (size_t)(j0 + j) * HD + c * 8),
                (ASL v4i*)&lds_k[j * 72 + c * 8], 0, 0);
        }
#else
        #pragma unroll
        for (int e = 0; e < 8; ++e) {
            int idx = t + e * 256;                 // 0..2047 (dwords)
            int j = idx >> 5, c = idx & 31;
            *(uint32_t*)&lds_k[j * 72 + c * 2] =
                *(const uint32_t*)(k_ws + base + (size_t)(j0 + j) * HD + c * 2);
        }
#endif
        // ---- stage V tile transposed [d][j]: b128 gathers, then b16 stores ----
        {
            union { u32x4 u; _Float16 f[8]; } vv[2];
            int vj[2], vd[2];
            #pragma unroll
            for (int e = 0; e < 2; ++e) {
                int idx = t + e * 256;             // 0..511 (16B = 8 f16 chunks)
                vj[e] = idx >> 3;                  // key row j
                vd[e] = (idx & 7) * 8;             // d base
                vv[e].u = *(const u32x4*)(v_ws + base +
                                          (size_t)(j0 + vj[e]) * HD + vd[e]);
            }
            #pragma unroll
            for (int e = 0; e < 2; ++e)
                #pragma unroll
                for (int q = 0; q < 8; ++q)
                    lds_v[(vd[e] + q) * 72 + vj[e]] = vv[e].f[q];
        }
#if defined(HAVE_ASYNC)
        __builtin_amdgcn_s_wait_asynccnt(0);
#endif
        __syncthreads();

        // S = Q * K^T  (B fragment: k-dim = d, n = key j; contiguous in d)
        v8f sf[4] = {};
        #pragma unroll
        for (int ks = 0; ks < 2; ++ks) {
            #pragma unroll
            for (int nt = 0; nt < 4; ++nt) {
                v16h bf;
                const _Float16* bp =
                    &lds_k[(nt * 16 + l16) * 72 + ks * 32 + hlf * 16];
                #pragma unroll
                for (int i = 0; i < 16; ++i) bf[i] = bp[i];
                sf[nt] = wmma_f16(aq[ks], bf, sf[nt]);
            }
        }

        // online softmax: rows live in 16-lane groups -> xor-butterfly 1,2,4,8
        float alpha[8];
        #pragma unroll
        for (int i = 0; i < 8; ++i) {
            float rmax = fmaxf(fmaxf(sf[0][i], sf[1][i]),
                               fmaxf(sf[2][i], sf[3][i]));
            #pragma unroll
            for (int mm = 1; mm < 16; mm <<= 1)
                rmax = fmaxf(rmax, __shfl_xor(rmax, mm, 32));
            const float mn = fmaxf(m[i], rmax);
            alpha[i] = __expf(m[i] - mn);
            m[i] = mn;
            float rs = 0.0f;
            #pragma unroll
            for (int nt = 0; nt < 4; ++nt) {
                float p = __expf(sf[nt][i] - mn);
                sf[nt][i] = p;
                rs += p;
            }
            #pragma unroll
            for (int mm = 1; mm < 16; mm <<= 1)
                rs += __shfl_xor(rs, mm, 32);
            l[i] = l[i] * alpha[i] + rs;
        }
        #pragma unroll
        for (int nt = 0; nt < 4; ++nt)
            #pragma unroll
            for (int i = 0; i < 8; ++i)
                o[nt][i] *= alpha[i];

        // stage P (f16) in this wave's private LDS region, then reload as A
        _Float16* pp = &lds_p[wave * (16 * 72)];
        #pragma unroll
        for (int nt = 0; nt < 4; ++nt)
            #pragma unroll
            for (int i = 0; i < 8; ++i)
                pp[(hlf * 8 + i) * 72 + nt * 16 + l16] = (_Float16)sf[nt][i];
        // intra-wave cross-lane LDS RAW: wait for DS stores before reload
        asm volatile("s_wait_dscnt 0" ::: "memory");

        // O += P * V   (A from lds_p, B from transposed lds_v: contiguous j)
        #pragma unroll
        for (int ks = 0; ks < 2; ++ks) {
            v16h ap;
            {
                const _Float16* src = &pp[l16 * 72 + ks * 32];
                const int klo = hlf * 16;
                #pragma unroll
                for (int i = 0; i < 8; ++i) ap[i]     = src[klo + i];
                #pragma unroll
                for (int i = 0; i < 8; ++i) ap[8 + i] = src[klo + 8 + i];
            }
            #pragma unroll
            for (int nt = 0; nt < 4; ++nt) {
                v16h bf;
                const _Float16* bp =
                    &lds_v[(nt * 16 + l16) * 72 + ks * 32 + hlf * 16];
                #pragma unroll
                for (int i = 0; i < 16; ++i) bf[i] = bp[i];
                o[nt] = wmma_f16(ap, bf, o[nt]);
            }
        }
        __syncthreads();
    }

    // epilogue: normalize by row sum, store fp32 in [B, S, H*64] layout
    #pragma unroll
    for (int nt = 0; nt < 4; ++nt) {
        #pragma unroll
        for (int i = 0; i < 8; ++i) {
            int M = hlf * 8 + i;
            int s = q0 + wave * 16 + M;
            float val = o[nt][i] / l[i];
            out[((size_t)(b * SEQ + s) * D_MODEL) + h * HD + nt * 16 + l16] = val;
        }
    }
}

// ---------------------------------------------------------------------------
extern "C" void kernel_launch(void* const* d_in, const int* in_sizes, int n_in,
                              void* d_out, int out_size, void* d_ws, size_t ws_size,
                              hipStream_t stream) {
    (void)in_sizes; (void)n_in; (void)out_size; (void)ws_size;
    const float* x  = (const float*)d_in[0];
    const float* Wq = (const float*)d_in[1];
    const float* bq = (const float*)d_in[2];
    const float* Wk = (const float*)d_in[3];
    const float* bk = (const float*)d_in[4];
    const float* Wv = (const float*)d_in[5];
    const float* bv = (const float*)d_in[6];
    float* out = (float*)d_out;

    const size_t per = (size_t)BATCH * N_HEADS * SEQ * HD;  // 4M f16 = 8 MB
    _Float16* q_ws = (_Float16*)d_ws;
    _Float16* k_ws = q_ws + per;
    _Float16* v_ws = k_ws + per;

    dim3 gproj(N_HEADS, (BATCH * SEQ) / 128, 3);
    qkv_proj_kernel<<<gproj, 256, 0, stream>>>(x, Wq, bq, Wk, bk, Wv, bv,
                                               q_ws, k_ws, v_ws);

    dim3 gattn(SEQ / 128, BATCH * N_HEADS);
    attn_kernel<<<gattn, 256, 0, stream>>>(q_ws, k_ws, v_ws, out);
}
